// GSNormDecoder_9526237462807
// MI455X (gfx1250) — matrix operand order repaired and checked by reading, not compiled, not run on hardware
//
#include <hip/hip_runtime.h>

// ---------------------------------------------------------------------------
// GSNormDecoder for MI455X (gfx1250): WMMA-f16 GEMM core + fp32 epilogues.
// All GEMMs are exact-tiled (M%64==0, N%256==0, K%32==0) -> no bounds guards.
// ---------------------------------------------------------------------------

typedef _Float16 v16h __attribute__((ext_vector_type(16)));
typedef float    v8f  __attribute__((ext_vector_type(8)));

#define B_   16
#define S_   512
#define L_   511
#define D_   1024
#define HID_ 512
#define DFF_ 2048
#define EA_  256
#define NH_  4
#define HD_  256
#define BH_  64   // B_*NH_

// ------------------------------ WMMA GEMM ----------------------------------
// C[bz][M,N] = act( A[bz][M,K] * B[bz][N,K]^T + bias ),  ldb == K always.
// block tile 64x256 (8 waves, each 32x64 = 2x4 WMMA tiles), K-slice 32,
// register-staged global->LDS pipeline with double-buffered LDS.
#define TM 64
#define TN 256
#define TK 32
#define LDH 40   // padded LDS row stride in halfs (32 data + 8 pad)

union AFrag { v16h v; _Float16 h[16]; };
union CFrag { v8f  v; float    f[8];  };

__device__ __forceinline__ float4 ld4(const float* p) {
    return *(const float4*)p;
}
__device__ __forceinline__ void st4h(_Float16* p, float4 v) {
    union { _Float16 h[4]; unsigned long long u; } t;
    t.h[0] = (_Float16)v.x; t.h[1] = (_Float16)v.y;
    t.h[2] = (_Float16)v.z; t.h[3] = (_Float16)v.w;
    *(unsigned long long*)p = t.u;   // 8B aligned: col offset multiple of 4 halfs
}

__global__ __launch_bounds__(256)
void gemm_wmma_kernel(const float* __restrict__ A, const float* __restrict__ Bm,
                      float* __restrict__ C, const float* __restrict__ bias,
                      int M, int N, int K,
                      long long sA, long long sB, long long sC, int act)
{
    __shared__ _Float16 Ash[2][TM * LDH];   //  2 x  5120 B
    __shared__ _Float16 Bsh[2][TN * LDH];   //  2 x 20480 B

    const int tid  = threadIdx.x;
    const int lane = tid & 31;
    const int wave = tid >> 5;
    const int wr   = wave >> 2;   // 0..1 : wave row (32 rows each)
    const int wc   = wave & 3;    // 0..3 : wave col (64 cols each)

    const float* Ab = A  + (long long)blockIdx.z * sA;
    const float* Bb = Bm + (long long)blockIdx.z * sB;
    float*       Cb = C  + (long long)blockIdx.z * sC;

    const int rowBase = blockIdx.y * TM;
    const int colBase = blockIdx.x * TN;

    // staging map: thread -> (row r0 + 32*i, 4 cols at c0)
    const int r0 = tid >> 3;          // 0..31
    const int c0 = (tid & 7) * 4;     // 0,4,...,28
    const float* aP0 = Ab + (long long)(rowBase + r0) * K + c0;
    const float* aP1 = aP0 + 32LL * K;
    const float* bP  = Bb + (long long)(colBase + r0) * K + c0;
    const long long bRow32 = 32LL * K;

    CFrag acc[2][4];
#pragma unroll
    for (int t = 0; t < 2; ++t)
#pragma unroll
        for (int u = 0; u < 4; ++u)
            acc[t][u].v = (v8f){0.f, 0.f, 0.f, 0.f, 0.f, 0.f, 0.f, 0.f};

    const int ml  = lane & 15;
    const int hiA = (lane >> 4) << 3;   // A frag K-half select (ISA 7.12.2)
    const int hiB = (lane >> 4) << 4;   // unused: B frags also use hiA (same k order as A)

    const int nk = K / TK;

    // prologue: stage slice 0 into registers
    float4 aR[2], bR[8];
    aR[0] = ld4(aP0); aR[1] = ld4(aP1);
#pragma unroll
    for (int i = 0; i < 8; ++i) bR[i] = ld4(bP + bRow32 * i);

    for (int kt = 0; kt < nk; ++kt) {
        const int buf = kt & 1;
        // regs -> LDS (f32 -> f16)
        st4h(&Ash[buf][r0 * LDH + c0],        aR[0]);
        st4h(&Ash[buf][(r0 + 32) * LDH + c0], aR[1]);
#pragma unroll
        for (int i = 0; i < 8; ++i)
            st4h(&Bsh[buf][(r0 + 32 * i) * LDH + c0], bR[i]);
        __syncthreads();

        // issue next slice's global loads (hidden behind WMMA below)
        if (kt + 1 < nk) {
            const int ko = (kt + 1) * TK;
            aR[0] = ld4(aP0 + ko); aR[1] = ld4(aP1 + ko);
#pragma unroll
            for (int i = 0; i < 8; ++i) bR[i] = ld4(bP + bRow32 * i + ko);
        }
        if (kt + 2 < nk)
            __builtin_prefetch(aP0 + (kt + 2) * TK, 0, 0);  // global_prefetch

        // fragments from LDS
        AFrag af[2], bf[4];
#pragma unroll
        for (int t = 0; t < 2; ++t) {
            const _Float16* ap = &Ash[buf][(wr * 32 + t * 16 + ml) * LDH];
#pragma unroll
            for (int i = 0; i < 16; ++i)
                af[t].h[i] = ap[hiA + i + ((i >= 8) ? 8 : 0)];
        }
#pragma unroll
        for (int u = 0; u < 4; ++u) {
            const _Float16* bp = &Bsh[buf][(wc * 64 + u * 16 + ml) * LDH];
#pragma unroll
            for (int i = 0; i < 16; ++i)
                bf[u].h[i] = bp[hiA + i + ((i >= 8) ? 8 : 0)];
        }
        // 8x v_wmma_f32_16x16x32_f16
#pragma unroll
        for (int t = 0; t < 2; ++t)
#pragma unroll
            for (int u = 0; u < 4; ++u)
                acc[t][u].v = __builtin_amdgcn_wmma_f32_16x16x32_f16(
                    false, af[t].v, false, bf[u].v,
                    (short)0, acc[t][u].v, false, false);
        // single barrier per slice: next iter writes the other LDS buffer
    }

    // ---- epilogue: bias + activation, fp32 stores (exact tiles) ----
#pragma unroll
    for (int u = 0; u < 4; ++u) {
        const int n = colBase + wc * 64 + u * 16 + ml;
        const float bv = bias ? bias[n] : 0.f;
#pragma unroll
        for (int t = 0; t < 2; ++t) {
            const int mb = rowBase + wr * 32 + t * 16 + hiA;
#pragma unroll
            for (int e = 0; e < 8; ++e) {
                float v = acc[t][u].f[e] + bv;
                if (act == 1)      v = v > 0.f ? v : 0.f;        // relu
                else if (act == 2) v = v >= 0.f ? v : 0.01f * v; // leaky
                Cb[(long long)(mb + e) * N + n] = v;
            }
        }
    }
}

// --------------------------- elementwise kernels ---------------------------

__global__ void build_tgt_kernel(const float* __restrict__ target,
                                 const float* __restrict__ bos,
                                 const float* __restrict__ w1, const float* __restrict__ b1,
                                 const float* __restrict__ w2, const float* __restrict__ b2,
                                 const float* __restrict__ prev, const float* __restrict__ cur,
                                 float* __restrict__ tgt)
{
    const int row = blockIdx.x;              // b*S + s
    const int s = row & (S_ - 1), b = row >> 9;
    const int tid = threadIdx.x;
    const float t = (s == 0) ? bos[0] : target[b * L_ + (s - 1)];

    __shared__ float h1[HID_];
#pragma unroll
    for (int i = 0; i < 2; ++i) {
        int j = tid + i * 256;
        float v = t * w1[j] + b1[j];
        h1[j] = (v >= 0.f) ? v : 0.01f * v;
    }
    __syncthreads();

    float* outr = tgt + (long long)row * D_;
#pragma unroll
    for (int i = 0; i < 2; ++i) {
        int j = tid + i * 256;
        float acc = b2[j];
        for (int kk = 0; kk < HID_; ++kk) acc += h1[kk] * w2[j * HID_ + kk];
        float ex  = -((float)((j >> 1) << 1)) * (9.2103403719761836f / (float)HID_);
        float ang = (float)s * __expf(ex);
        acc += (j & 1) ? __cosf(ang) : __sinf(ang);
        outr[j] = acc;
    }
    outr[HID_ + tid]       = prev[b * EA_ + tid];   // EA_ == 256 == blockDim
    outr[HID_ + EA_ + tid] = cur[b * EA_ + tid];
}

__global__ void build_mem_kernel(const float* __restrict__ gs,
                                 const float* __restrict__ prev,
                                 const float* __restrict__ cur,
                                 float* __restrict__ mem)
{
    const long long total = (long long)B_ * S_ * D_;
    for (long long idx = (long long)blockIdx.x * blockDim.x + threadIdx.x;
         idx < total; idx += (long long)gridDim.x * blockDim.x) {
        int j = (int)(idx & (D_ - 1));
        long long row = idx >> 10;
        int b = (int)(row >> 9);
        float v;
        if (j < HID_)            v = gs[row * HID_ + j];
        else if (j < HID_ + EA_) v = prev[b * EA_ + (j - HID_)];
        else                     v = cur[b * EA_ + (j - HID_ - EA_)];
        mem[idx] = v;
    }
}

// src [B*S, srcCols] slice -> dst [B, NH, S, HD]
__global__ void split_heads_kernel(const float* __restrict__ src, int srcCols, int colOff,
                                   float* __restrict__ dst)
{
    const long long total = (long long)B_ * S_ * D_;
    for (long long idx = (long long)blockIdx.x * blockDim.x + threadIdx.x;
         idx < total; idx += (long long)gridDim.x * blockDim.x) {
        int n = (int)(idx & (D_ - 1));
        long long row = idx >> 10;
        int s = (int)(row & (S_ - 1)), b = (int)(row >> 9);
        int h = n >> 8, d = n & (HD_ - 1);
        dst[(((long long)(b * NH_ + h) * S_) + s) * HD_ + d] =
            src[row * srcCols + colOff + n];
    }
}

// src [B*S, srcCols] slice -> dst [B, NH, HD, S]   (head-transposed, for V)
__global__ void split_heads_t_kernel(const float* __restrict__ src, int srcCols, int colOff,
                                     float* __restrict__ dst)
{
    const long long total = (long long)B_ * S_ * D_;
    for (long long idx = (long long)blockIdx.x * blockDim.x + threadIdx.x;
         idx < total; idx += (long long)gridDim.x * blockDim.x) {
        int n = (int)(idx & (D_ - 1));
        long long row = idx >> 10;
        int s = (int)(row & (S_ - 1)), b = (int)(row >> 9);
        int h = n >> 8, d = n & (HD_ - 1);
        dst[(((long long)(b * NH_ + h) * HD_) + d) * S_ + s] =
            src[row * srcCols + colOff + n];
    }
}

// src [B, NH, S, HD] -> dst [B*S, D]
__global__ void merge_heads_kernel(const float* __restrict__ src, float* __restrict__ dst)
{
    const long long total = (long long)B_ * S_ * D_;
    for (long long idx = (long long)blockIdx.x * blockDim.x + threadIdx.x;
         idx < total; idx += (long long)gridDim.x * blockDim.x) {
        int n = (int)(idx & (D_ - 1));
        long long row = idx >> 10;
        int s = (int)(row & (S_ - 1)), b = (int)(row >> 9);
        int h = n >> 8, d = n & (HD_ - 1);
        dst[idx] = src[(((long long)(b * NH_ + h) * S_) + s) * HD_ + d];
    }
}

// scores [BH, S, S] rows; optional causal mask; scale = 1/sqrt(hd)
__global__ __launch_bounds__(256)
void softmax_rows_kernel(float* __restrict__ sc, int causal, float scale)
{
    const int gr = blockIdx.x;          // bh*S + q
    const int qrow = gr & (S_ - 1);
    float* row = sc + (long long)gr * S_;
    const int tid = threadIdx.x;
    __shared__ float red[256];

    bool m0 = causal && (tid > qrow);
    bool m1 = causal && (tid + 256 > qrow);
    float x0 = m0 ? -1e30f : row[tid] * scale;
    float x1 = m1 ? -1e30f : row[tid + 256] * scale;

    red[tid] = fmaxf(x0, x1); __syncthreads();
    for (int s = 128; s > 0; s >>= 1) {
        if (tid < s) red[tid] = fmaxf(red[tid], red[tid + s]);
        __syncthreads();
    }
    float mx = red[0]; __syncthreads();

    float e0 = m0 ? 0.f : __expf(x0 - mx);
    float e1 = m1 ? 0.f : __expf(x1 - mx);
    red[tid] = e0 + e1; __syncthreads();
    for (int s = 128; s > 0; s >>= 1) {
        if (tid < s) red[tid] += red[tid + s];
        __syncthreads();
    }
    float inv = 1.f / red[0];
    row[tid] = e0 * inv;
    row[tid + 256] = e1 * inv;
}

// h = LayerNorm(h + y) * w + b, row length D
__global__ __launch_bounds__(256)
void add_ln_kernel(float* __restrict__ h, const float* __restrict__ y,
                   const float* __restrict__ w, const float* __restrict__ b)
{
    const long long row = blockIdx.x;
    float* hr = h + row * D_;
    const float* yr = y + row * D_;
    const int tid = threadIdx.x;
    float x[4], s1 = 0.f, s2 = 0.f;
#pragma unroll
    for (int i = 0; i < 4; ++i) {
        int j = tid + i * 256;
        float v = hr[j] + yr[j];
        x[i] = v; s1 += v; s2 += v * v;
    }
    __shared__ float r1[256], r2[256];
    r1[tid] = s1; r2[tid] = s2; __syncthreads();
    for (int s = 128; s > 0; s >>= 1) {
        if (tid < s) { r1[tid] += r1[tid + s]; r2[tid] += r2[tid + s]; }
        __syncthreads();
    }
    float mean = r1[0] * (1.f / D_);
    float var  = r2[0] * (1.f / D_) - mean * mean;
    float inv  = rsqrtf(var + 1e-5f);
#pragma unroll
    for (int i = 0; i < 4; ++i) {
        int j = tid + i * 256;
        hr[j] = (x[i] - mean) * inv * w[j] + b[j];
    }
}

// out[b,l] = mask * ( ph[b*S + l + 1, :] . pw2 + pb2 )
__global__ __launch_bounds__(256)
void final_out_kernel(const float* __restrict__ ph, const float* __restrict__ pw2,
                      const float* __restrict__ pb2, const int* __restrict__ basis,
                      float* __restrict__ out)
{
    const int idx = blockIdx.x;         // b*L + l
    const int b = idx / L_, l = idx % L_;
    const float* r = ph + (long long)(b * S_ + l + 1) * HID_;
    const int tid = threadIdx.x;
    __shared__ float red[256];
    red[tid] = r[tid] * pw2[tid] + r[tid + 256] * pw2[tid + 256];
    __syncthreads();
    for (int s = 128; s > 0; s >>= 1) {
        if (tid < s) red[tid] += red[tid + s];
        __syncthreads();
    }
    if (tid == 0) {
        float v = red[0] + pb2[0];
        out[idx] = (l < basis[b]) ? v : 0.f;
    }
}

// ------------------------------- host side ---------------------------------

static inline void launch_gemm(hipStream_t st, const float* A, const float* Bm, float* C,
                               const float* bias, int M, int N, int K, int batch,
                               long long sA, long long sB, long long sC, int act)
{
    dim3 grid(N / TN, M / TM, batch);
    gemm_wmma_kernel<<<grid, dim3(256), 0, st>>>(A, Bm, C, bias, M, N, K, sA, sB, sC, act);
}

extern "C" void kernel_launch(void* const* d_in, const int* in_sizes, int n_in,
                              void* d_out, int out_size, void* d_ws, size_t ws_size,
                              hipStream_t stream)
{
    (void)in_sizes; (void)n_in; (void)out_size; (void)ws_size;

    const float* gs       = (const float*)d_in[0];
    const float* prevA    = (const float*)d_in[1];
    const float* curA     = (const float*)d_in[2];
    const float* tnorms   = (const float*)d_in[3];
    const float* bos      = (const float*)d_in[4];
    const float* inp_w1   = (const float*)d_in[5];
    const float* inp_b1   = (const float*)d_in[6];
    const float* inp_w2   = (const float*)d_in[7];
    const float* inp_b2   = (const float*)d_in[8];
    const float* sa_qkv_w = (const float*)d_in[9];
    const float* sa_qkv_b = (const float*)d_in[10];
    const float* sa_out_w = (const float*)d_in[11];
    const float* sa_out_b = (const float*)d_in[12];
    const float* ca_qkv_w = (const float*)d_in[13];
    const float* ca_qkv_b = (const float*)d_in[14];
    const float* ca_out_w = (const float*)d_in[15];
    const float* ca_out_b = (const float*)d_in[16];
    const float* ln_w     = (const float*)d_in[17];
    const float* ln_b     = (const float*)d_in[18];
    const float* ff_w1    = (const float*)d_in[19];
    const float* ff_b1    = (const float*)d_in[20];
    const float* ff_w2    = (const float*)d_in[21];
    const float* ff_b2    = (const float*)d_in[22];
    const float* proj_w1  = (const float*)d_in[23];
    const float* proj_b1  = (const float*)d_in[24];
    const float* proj_w2  = (const float*)d_in[25];
    const float* proj_b2  = (const float*)d_in[26];
    const int*   basis    = (const int*)d_in[27];
    float* out = (float*)d_out;

    // -------- workspace partition (floats) --------
    const long long ROWS = (long long)B_ * S_;          // 8192
    float* ws = (float*)d_ws;
    long long o = 0;
    float* h    = ws + o; o += ROWS * D_;               // residual stream
    float* mem  = ws + o; o += ROWS * D_;               // encoder memory
    float* big  = ws + o; o += ROWS * 3 * D_;           // qkv / kv / ffn-hidden / merged
    float* qb   = ws + o; o += ROWS * D_;               // [B,NH,S,HD]
    float* kb   = ws + o; o += ROWS * D_;               // [B,NH,S,HD]
    float* vb   = ws + o; o += ROWS * D_;               // [B,NH,HD,S]  (transposed)
    float* sc   = ws + o; o += (long long)BH_ * S_ * S_;// attention scores
    float* ao   = ws + o; o += ROWS * D_;               // per-head attn out
    float* tmp  = ws + o; o += ROWS * D_;               // sublayer output before LN
    float* ph   = ws + o; o += ROWS * HID_;             // final projection hidden

    const int EW = 256;
    const int EG = 32768;
    const float iscale = 0.0625f;           // 1/sqrt(256)

    build_tgt_kernel<<<dim3(B_ * S_), dim3(EW), 0, stream>>>(
        tnorms, bos, inp_w1, inp_b1, inp_w2, inp_b2, prevA, curA, h);
    build_mem_kernel<<<dim3(EG), dim3(EW), 0, stream>>>(gs, prevA, curA, mem);

    for (int l = 0; l < 3; ++l) {
        const float* qkvW  = sa_qkv_w + (long long)l * 3 * D_ * D_;
        const float* qkvB  = sa_qkv_b + (long long)l * 3 * D_;
        const float* outW  = sa_out_w + (long long)l * D_ * D_;
        const float* outB  = sa_out_b + (long long)l * D_;
        const float* cqkvW = ca_qkv_w + (long long)l * 3 * D_ * D_;
        const float* cqkvB = ca_qkv_b + (long long)l * 3 * D_;
        const float* coutW = ca_out_w + (long long)l * D_ * D_;
        const float* coutB = ca_out_b + (long long)l * D_;
        const float* lw = ln_w + (long long)l * 3 * D_;
        const float* lb = ln_b + (long long)l * 3 * D_;

        // ---- self attention ----
        launch_gemm(stream, h, qkvW, big, qkvB, (int)ROWS, 3 * D_, D_, 1, 0, 0, 0, 0);
        split_heads_kernel  <<<dim3(EG), dim3(EW), 0, stream>>>(big, 3 * D_, 0,      qb);
        split_heads_kernel  <<<dim3(EG), dim3(EW), 0, stream>>>(big, 3 * D_, D_,     kb);
        split_heads_t_kernel<<<dim3(EG), dim3(EW), 0, stream>>>(big, 3 * D_, 2 * D_, vb);
        launch_gemm(stream, qb, kb, sc, nullptr, S_, S_, HD_, BH_,
                    (long long)S_ * HD_, (long long)S_ * HD_, (long long)S_ * S_, 0);
        softmax_rows_kernel<<<dim3(BH_ * S_), dim3(EW), 0, stream>>>(sc, 1, iscale);
        launch_gemm(stream, sc, vb, ao, nullptr, S_, HD_, S_, BH_,
                    (long long)S_ * S_, (long long)HD_ * S_, (long long)S_ * HD_, 0);
        merge_heads_kernel<<<dim3(EG), dim3(EW), 0, stream>>>(ao, big);
        launch_gemm(stream, big, outW, tmp, outB, (int)ROWS, D_, D_, 1, 0, 0, 0, 0);
        add_ln_kernel<<<dim3((int)ROWS), dim3(EW), 0, stream>>>(h, tmp, lw, lb);

        // ---- cross attention (q from h, k/v from mem) ----
        launch_gemm(stream, h, cqkvW, big, cqkvB, (int)ROWS, D_, D_, 1, 0, 0, 0, 0);
        launch_gemm(stream, mem, cqkvW + (long long)D_ * D_, big + ROWS * D_,
                    cqkvB + D_, (int)ROWS, 2 * D_, D_, 1, 0, 0, 0, 0);
        split_heads_kernel  <<<dim3(EG), dim3(EW), 0, stream>>>(big, D_, 0, qb);
        split_heads_kernel  <<<dim3(EG), dim3(EW), 0, stream>>>(big + ROWS * D_, 2 * D_, 0,  kb);
        split_heads_t_kernel<<<dim3(EG), dim3(EW), 0, stream>>>(big + ROWS * D_, 2 * D_, D_, vb);
        launch_gemm(stream, qb, kb, sc, nullptr, S_, S_, HD_, BH_,
                    (long long)S_ * HD_, (long long)S_ * HD_, (long long)S_ * S_, 0);
        softmax_rows_kernel<<<dim3(BH_ * S_), dim3(EW), 0, stream>>>(sc, 0, iscale);
        launch_gemm(stream, sc, vb, ao, nullptr, S_, HD_, S_, BH_,
                    (long long)S_ * S_, (long long)HD_ * S_, (long long)S_ * HD_, 0);
        merge_heads_kernel<<<dim3(EG), dim3(EW), 0, stream>>>(ao, big);
        launch_gemm(stream, big, coutW, tmp, coutB, (int)ROWS, D_, D_, 1, 0, 0, 0, 0);
        add_ln_kernel<<<dim3((int)ROWS), dim3(EW), 0, stream>>>(h, tmp, lw + D_, lb + D_);

        // ---- feed-forward ----
        launch_gemm(stream, h, ff_w1 + (long long)l * DFF_ * D_, big,
                    ff_b1 + (long long)l * DFF_, (int)ROWS, DFF_, D_, 1, 0, 0, 0, 1);
        launch_gemm(stream, big, ff_w2 + (long long)l * D_ * DFF_, tmp,
                    ff_b2 + (long long)l * D_, (int)ROWS, D_, DFF_, 1, 0, 0, 0, 0);
        add_ln_kernel<<<dim3((int)ROWS), dim3(EW), 0, stream>>>(h, tmp, lw + 2 * D_, lb + 2 * D_);
    }

    launch_gemm(stream, h, proj_w1, ph, proj_b1, (int)ROWS, HID_, D_, 1, 0, 0, 0, 2);
    final_out_kernel<<<dim3(B_ * L_), dim3(EW), 0, stream>>>(ph, proj_w2, proj_b2, basis, out);
}
